// HungarianMatcher_41274635715334
// MI455X (gfx1250) — compile-verified, matrix operand order checked by reading, not run on hardware
//
#include <hip/hip_runtime.h>
#include <math.h>

// Problem constants (from the reference)
#define BSZ 8
#define NQ  100
#define NC  81
#define MTG 160           // number of targets
#define PPX 65536         // pixels per mask (256*256)
#define NPR (BSZ * NQ)    // 800 predictions

#define HDR 4096          // floats reserved at start of workspace for row stats

#define NTIL 5            // m-tiles per wave (16 targets each)
#define NMB  2            // m-blocks (NTIL * NMB * 16 == MTG)

#define LOG2E 1.4426950408889634f
#define LN2   0.6931471805599453f

typedef __attribute__((ext_vector_type(16))) __bf16 v16bf;
typedef __attribute__((ext_vector_type(8)))  __bf16 v8bf;
typedef __attribute__((ext_vector_type(8)))  float  v8f;

// Fast sigmoid: one v_exp_f32 + one v_rcp_f32 (no IEEE divide expansion).
__device__ __forceinline__ float fast_sigmoid(float x) {
  return __builtin_amdgcn_rcpf(1.f + __builtin_amdgcn_exp2f(-x * LOG2E));
}

// Load 16 bf16 laid out as {p[0..7], p[16..23]} into a v16bf (WMMA K pattern).
__device__ __forceinline__ v16bf load_b16(const __bf16* p) {
  v8bf lo = *reinterpret_cast<const v8bf*>(p);
  v8bf hi = *reinterpret_cast<const v8bf*>(p + 16);
  return __builtin_shufflevector(lo, hi, 0, 1, 2, 3, 4, 5, 6, 7,
                                 8, 9, 10, 11, 12, 13, 14, 15);
}

// ---------------------------------------------------------------------------
// Kernel 1: per-prediction row stats over P pixels:
//   rowPos[n] = sum_p max(x,0) + log1p(exp(-|x|))   (BCE row term)
//   rowS[n]   = sum_p sigmoid(x)                    (dice denominator term)
// softplus via native log2/exp2: max(x,0) + ln2*log2(1 + 2^(-|x|*log2e))
// ---------------------------------------------------------------------------
__global__ void row_stats_kernel(const float* __restrict__ x,
                                 float* __restrict__ rowPos,
                                 float* __restrict__ rowS) {
  __shared__ float sp[256];
  __shared__ float ss[256];
  const int n = blockIdx.x;
  const int tid = threadIdx.x;
  const float* row = x + (size_t)n * PPX;
  float accp = 0.f, accs = 0.f;
  for (int i = tid * 4; i < PPX; i += 256 * 4) {
    float4 v = *reinterpret_cast<const float4*>(row + i);
    float a[4] = {v.x, v.y, v.z, v.w};
#pragma unroll
    for (int j = 0; j < 4; ++j) {
      float xv = a[j];
      float e = __builtin_amdgcn_exp2f(-fabsf(xv) * LOG2E);
      accp += fmaxf(xv, 0.f) + LN2 * __builtin_amdgcn_logf(1.f + e);
      accs += fast_sigmoid(xv);
    }
  }
  sp[tid] = accp; ss[tid] = accs;
  __syncthreads();
  for (int s = 128; s > 0; s >>= 1) {
    if (tid < s) { sp[tid] += sp[tid + s]; ss[tid] += ss[tid + s]; }
    __syncthreads();
  }
  if (tid == 0) { rowPos[n] = sp[0]; rowS[n] = ss[0]; }
}

// ---------------------------------------------------------------------------
// Kernel 2a: fused target conversion + row sum.  t is binary so f32->bf16 is
// exact; the 21MB bf16 copy stays resident in the 192MB L2 for the GEMM.
// ---------------------------------------------------------------------------
__global__ void tconv_kernel(const float* __restrict__ t,
                             __bf16* __restrict__ tb,
                             float* __restrict__ tSum) {
  __shared__ float sp[256];
  const int m = blockIdx.x;
  const int tid = threadIdx.x;
  const float* row = t + (size_t)m * PPX;
  __bf16* orow = tb + (size_t)m * PPX;
  float acc = 0.f;
  for (int i = tid * 8; i < PPX; i += 256 * 8) {
    float4 v0 = *reinterpret_cast<const float4*>(row + i);
    float4 v1 = *reinterpret_cast<const float4*>(row + i + 4);
    acc += (v0.x + v0.y) + (v0.z + v0.w) + (v1.x + v1.y) + (v1.z + v1.w);
    v8bf o;
    o[0] = (__bf16)v0.x; o[1] = (__bf16)v0.y; o[2] = (__bf16)v0.z; o[3] = (__bf16)v0.w;
    o[4] = (__bf16)v1.x; o[5] = (__bf16)v1.y; o[6] = (__bf16)v1.z; o[7] = (__bf16)v1.w;
    *reinterpret_cast<v8bf*>(orow + i) = o;
  }
  sp[tid] = acc;
  __syncthreads();
  for (int s = 128; s > 0; s >>= 1) {
    if (tid < s) sp[tid] += sp[tid + s];
    __syncthreads();
  }
  if (tid == 0) tSum[m] = sp[0];
}

// Kernel 2b (fallback, tiny workspace): row sums only.
__global__ void tsum_kernel(const float* __restrict__ t,
                            float* __restrict__ tSum) {
  __shared__ float sp[256];
  const int m = blockIdx.x;
  const int tid = threadIdx.x;
  const float* row = t + (size_t)m * PPX;
  float acc = 0.f;
  for (int i = tid * 4; i < PPX; i += 256 * 4) {
    float4 v = *reinterpret_cast<const float4*>(row + i);
    acc += v.x + v.y + v.z + v.w;
  }
  sp[tid] = acc;
  __syncthreads();
  for (int s = 128; s > 0; s >>= 1) {
    if (tid < s) sp[tid] += sp[tid + s];
    __syncthreads();
  }
  if (tid == 0) tSum[m] = sp[0];
}

// ---------------------------------------------------------------------------
// Kernel 3: softmax stats per prediction row (81 classes): max + denom
// ---------------------------------------------------------------------------
__global__ void class_stats_kernel(const float* __restrict__ logits,
                                   float* __restrict__ clsMax,
                                   float* __restrict__ clsDen) {
  const int n = blockIdx.x * blockDim.x + threadIdx.x;
  if (n >= NPR) return;
  const float* row = logits + (size_t)n * NC;
  float mx = row[0];
  for (int c = 1; c < NC; ++c) mx = fmaxf(mx, row[c]);
  float den = 0.f;
  for (int c = 0; c < NC; ++c) den += __expf(row[c] - mx);
  clsMax[n] = mx;
  clsDen[n] = den;
}

// ---------------------------------------------------------------------------
// Kernel 4: fused dual NT-GEMM via bf16 WMMA with hi/lo split-precision.
//   cross[n,m] = sum_p x[n,p] * t[m,p]
//   sdot [n,m] = sum_p sigmoid(x[n,p]) * t[m,p]
// One wave = 16 queries x 80 targets (NTIL=5 m-tiles); gridDim.z = 2 m-blocks;
// split-K over gridDim.y.  k-loop unrolled x2 so consecutive iterations live
// in distinct SSA values (no rotation v_movs, no WMMA->cvt WAR v_nops); B tile
// mt+1 is prefetched before tile mt's WMMAs; A stream prefetched ~1KB ahead.
// ---------------------------------------------------------------------------
__global__ __launch_bounds__(32)
void gemm_bf16_kernel(const float* __restrict__ x,
                      const __bf16* __restrict__ tb,
                      float* __restrict__ crossP,
                      float* __restrict__ sdotP) {
  const int lane = threadIdx.x;        // 0..31
  const int nb   = blockIdx.x;         // 0..49 query-block
  const int ks   = blockIdx.y;         // split-K slice
  const int mb   = blockIdx.z;         // 0..1 target-block
  const int KS   = gridDim.y;
  const int chunk = PPX / KS;          // KS is a power of two <= 32
  const int k0 = ks * chunk, kend = k0 + chunk;
  const int r15 = lane & 15;
  const int kh  = lane >> 4;
  const float*  xrow  = x  + (size_t)(nb * 16 + r15) * PPX;
  const __bf16* tbase = tb + (size_t)(mb * NTIL * 16 + r15) * PPX;

  v8f accC[NTIL], accS[NTIL];
#pragma unroll
  for (int i = 0; i < NTIL; ++i)
#pragma unroll
    for (int j = 0; j < 8; ++j) { accC[i][j] = 0.f; accS[i][j] = 0.f; }

#pragma unroll 2
  for (int k = k0; k < kend; k += 32) {
    const int ka = k + 8 * kh;
    // Speculative prefetch of the A (HBM) stream ~8 k-steps ahead.
    __builtin_prefetch(xrow + ka + 256, 0, 1);
    float4 a0 = *reinterpret_cast<const float4*>(xrow + ka);
    float4 a1 = *reinterpret_cast<const float4*>(xrow + ka + 4);
    float4 a2 = *reinterpret_cast<const float4*>(xrow + ka + 16);
    float4 a3 = *reinterpret_cast<const float4*>(xrow + ka + 20);
    float xv[16] = {a0.x, a0.y, a0.z, a0.w, a1.x, a1.y, a1.z, a1.w,
                    a2.x, a2.y, a2.z, a2.w, a3.x, a3.y, a3.z, a3.w};
    v16bf xh, xl, sh, sl;
#pragma unroll
    for (int i = 0; i < 16; ++i) {
      float xf = xv[i];
      __bf16 h = (__bf16)xf;              // hi part (RNE)
      xh[i] = h;
      xl[i] = (__bf16)(xf - (float)h);    // lo residual -> ~16 mantissa bits
      float s = fast_sigmoid(xf);
      __bf16 hs = (__bf16)s;
      sh[i] = hs;
      sl[i] = (__bf16)(s - (float)hs);
    }
    v16bf bcur = load_b16(tbase + ka);    // local m-tile 0
#pragma unroll
    for (int mt = 0; mt < NTIL; ++mt) {
      v16bf bnext = bcur;
      if (mt < NTIL - 1)
        bnext = load_b16(tbase + (size_t)(mt + 1) * 16 * PPX + ka);
      accC[mt] = __builtin_amdgcn_wmma_f32_16x16x32_bf16(
          false, xh, false, bcur, (short)0, accC[mt], false, false);
      accC[mt] = __builtin_amdgcn_wmma_f32_16x16x32_bf16(
          false, xl, false, bcur, (short)0, accC[mt], false, false);
      accS[mt] = __builtin_amdgcn_wmma_f32_16x16x32_bf16(
          false, sh, false, bcur, (short)0, accS[mt], false, false);
      accS[mt] = __builtin_amdgcn_wmma_f32_16x16x32_bf16(
          false, sl, false, bcur, (short)0, accS[mt], false, false);
      bcur = bnext;
    }
  }

  // C/D layout: VGPR j, lane l -> row = j + 8*(l>>4), col = l&15
  const size_t base = (size_t)ks * NPR * MTG;
#pragma unroll
  for (int mt = 0; mt < NTIL; ++mt) {
    const int gm = (mb * NTIL + mt) * 16 + r15;
#pragma unroll
    for (int j = 0; j < 8; ++j) {
      const int q = nb * 16 + j + 8 * kh;
      crossP[base + (size_t)q * MTG + gm] = accC[mt][j];
      sdotP [base + (size_t)q * MTG + gm] = accS[mt][j];
    }
  }
}

// ---------------------------------------------------------------------------
// Kernel 4-fallback (tiny workspace): same GEMM but converts f32 t on the fly.
// ---------------------------------------------------------------------------
__global__ __launch_bounds__(32)
void gemm_fused_kernel(const float* __restrict__ x,
                       const float* __restrict__ t,
                       float* __restrict__ crossP,
                       float* __restrict__ sdotP) {
  const int lane = threadIdx.x;
  const int nb   = blockIdx.x;
  const int ks   = blockIdx.y;
  const int mb   = blockIdx.z;
  const int KS   = gridDim.y;
  const int chunk = PPX / KS;
  const int k0 = ks * chunk;
  const int r15 = lane & 15;
  const int kh  = lane >> 4;
  const float* xrow = x + (size_t)(nb * 16 + r15) * PPX;

  v8f accC[NTIL], accS[NTIL];
#pragma unroll
  for (int i = 0; i < NTIL; ++i)
#pragma unroll
    for (int j = 0; j < 8; ++j) { accC[i][j] = 0.f; accS[i][j] = 0.f; }

  for (int k = k0; k < k0 + chunk; k += 32) {
    const int ka = k + 8 * kh;
    float4 a0 = *reinterpret_cast<const float4*>(xrow + ka);
    float4 a1 = *reinterpret_cast<const float4*>(xrow + ka + 4);
    float4 a2 = *reinterpret_cast<const float4*>(xrow + ka + 16);
    float4 a3 = *reinterpret_cast<const float4*>(xrow + ka + 20);
    float xv[16] = {a0.x, a0.y, a0.z, a0.w, a1.x, a1.y, a1.z, a1.w,
                    a2.x, a2.y, a2.z, a2.w, a3.x, a3.y, a3.z, a3.w};
    v16bf xh, xl, sh, sl;
#pragma unroll
    for (int i = 0; i < 16; ++i) {
      float xf = xv[i];
      __bf16 h = (__bf16)xf;
      xh[i] = h;
      xl[i] = (__bf16)(xf - (float)h);
      float s = fast_sigmoid(xf);
      __bf16 hs = (__bf16)s;
      sh[i] = hs;
      sl[i] = (__bf16)(s - (float)hs);
    }
#pragma unroll
    for (int mt = 0; mt < NTIL; ++mt) {
      const float* trow = t + (size_t)((mb * NTIL + mt) * 16 + r15) * PPX;
      float4 b0 = *reinterpret_cast<const float4*>(trow + ka);
      float4 b1 = *reinterpret_cast<const float4*>(trow + ka + 4);
      float4 b2 = *reinterpret_cast<const float4*>(trow + ka + 16);
      float4 b3 = *reinterpret_cast<const float4*>(trow + ka + 20);
      float bv[16] = {b0.x, b0.y, b0.z, b0.w, b1.x, b1.y, b1.z, b1.w,
                      b2.x, b2.y, b2.z, b2.w, b3.x, b3.y, b3.z, b3.w};
      v16bf tv;
#pragma unroll
      for (int i = 0; i < 16; ++i) tv[i] = (__bf16)bv[i];
      accC[mt] = __builtin_amdgcn_wmma_f32_16x16x32_bf16(
          false, xh, false, tv, (short)0, accC[mt], false, false);
      accC[mt] = __builtin_amdgcn_wmma_f32_16x16x32_bf16(
          false, xl, false, tv, (short)0, accC[mt], false, false);
      accS[mt] = __builtin_amdgcn_wmma_f32_16x16x32_bf16(
          false, sh, false, tv, (short)0, accS[mt], false, false);
      accS[mt] = __builtin_amdgcn_wmma_f32_16x16x32_bf16(
          false, sl, false, tv, (short)0, accS[mt], false, false);
    }
  }
  const size_t base = (size_t)ks * NPR * MTG;
#pragma unroll
  for (int mt = 0; mt < NTIL; ++mt) {
    const int gm = (mb * NTIL + mt) * 16 + r15;
#pragma unroll
    for (int j = 0; j < 8; ++j) {
      const int q = nb * 16 + j + 8 * kh;
      crossP[base + (size_t)q * MTG + gm] = accC[mt][j];
      sdotP [base + (size_t)q * MTG + gm] = accS[mt][j];
    }
  }
}

// ---------------------------------------------------------------------------
// Kernel 5: deterministic split-K reduction + final cost combine.
// ---------------------------------------------------------------------------
__global__ void combine_kernel(const float* __restrict__ logits,
                               const int* __restrict__ tgt_ids,
                               const float* __restrict__ crossP,
                               const float* __restrict__ sdotP,
                               const float* __restrict__ rowPos,
                               const float* __restrict__ rowS,
                               const float* __restrict__ clsMax,
                               const float* __restrict__ clsDen,
                               const float* __restrict__ tSum,
                               float* __restrict__ out, int KS) {
  const int idx = blockIdx.x * blockDim.x + threadIdx.x;
  if (idx >= NPR * MTG) return;
  const int n = idx / MTG;
  const int m = idx - n * MTG;
  float cr = 0.f, sd = 0.f;
  for (int ks = 0; ks < KS; ++ks) {
    const size_t o = ((size_t)ks * NPR + n) * MTG + m;
    cr += crossP[o];
    sd += sdotP[o];
  }
  const float cls = -__expf(logits[(size_t)n * NC + tgt_ids[m]] - clsMax[n]) / clsDen[n];
  const float cmask = (rowPos[n] - cr) * (1.0f / (float)PPX);
  const float cdice = 1.f - (2.f * sd + 1.f) / (rowS[n] + tSum[m] + 1.f);
  out[idx] = cls + cmask + cdice;   // all cost weights are 1.0
}

// ---------------------------------------------------------------------------
extern "C" void kernel_launch(void* const* d_in, const int* in_sizes, int n_in,
                              void* d_out, int out_size, void* d_ws, size_t ws_size,
                              hipStream_t stream) {
  const float* pred_logits = (const float*)d_in[0];  // [800, 81]
  const float* pred_masks  = (const float*)d_in[1];  // [800, 65536]
  const int*   tgt_ids     = (const int*)d_in[2];    // [160]
  const float* tgt_mask    = (const float*)d_in[3];  // [160, 65536]
  float* out = (float*)d_out;                        // [800, 160]
  float* ws  = (float*)d_ws;

  float* rowPos = ws;             // 800
  float* rowS   = ws + NPR;       // 800
  float* clsMax = ws + 2 * NPR;   // 800
  float* clsDen = ws + 3 * NPR;   // 800
  float* tSum   = ws + 4 * NPR;   // 160   (ends at 3360 < HDR=4096)

  const size_t TBF_F = (size_t)MTG * PPX / 2;   // bf16 t-copy, in float units
  const size_t slab  = (size_t)2 * NPR * MTG;   // cross+sdot per K-slice (floats)
  const size_t avail = ws_size / sizeof(float);

  const bool pre = avail >= HDR + TBF_F + slab; // room for bf16 copy + 1 slice?
  size_t rem = 0;
  if (pre)                 rem = avail - HDR - TBF_F;
  else if (avail > HDR)    rem = avail - HDR;
  int KS = 1;
  {
    size_t slots = rem / slab;
    int kmax = slots > 32 ? 32 : (int)slots;
    if (kmax < 1) kmax = 1;
    while (KS * 2 <= kmax) KS *= 2;
  }
  __bf16* tb    = (__bf16*)(ws + HDR);
  float* crossP = pre ? (ws + HDR + TBF_F) : (ws + HDR);
  float* sdotP  = crossP + (size_t)KS * NPR * MTG;

  row_stats_kernel<<<NPR, 256, 0, stream>>>(pred_masks, rowPos, rowS);
  class_stats_kernel<<<(NPR + 127) / 128, 128, 0, stream>>>(pred_logits, clsMax, clsDen);
  if (pre) {
    tconv_kernel<<<MTG, 256, 0, stream>>>(tgt_mask, tb, tSum);
    gemm_bf16_kernel<<<dim3(NPR / 16, KS, NMB), 32, 0, stream>>>(pred_masks, tb, crossP, sdotP);
  } else {
    tsum_kernel<<<MTG, 256, 0, stream>>>(tgt_mask, tSum);
    gemm_fused_kernel<<<dim3(NPR / 16, KS, NMB), 32, 0, stream>>>(pred_masks, tgt_mask, crossP, sdotP);
  }
  combine_kernel<<<(NPR * MTG + 255) / 256, 256, 0, stream>>>(
      pred_logits, tgt_ids, crossP, sdotP, rowPos, rowS, clsMax, clsDen, tSum, out, KS);
}
